// RNNTJoiner_83863531421968
// MI455X (gfx1250) — compile-verified, matrix operand order checked by reading, not run on hardware
//
#include <hip/hip_runtime.h>
#include <hip/hip_bf16.h>
#include <stdint.h>

typedef __attribute__((ext_vector_type(16))) _Float16     v16h;
typedef __attribute__((ext_vector_type(8)))  _Float16     v8h;
typedef __attribute__((ext_vector_type(8)))  float        v8f;
typedef __attribute__((ext_vector_type(4)))  unsigned int v4u;
typedef __attribute__((ext_vector_type(8)))  int          v8i;
typedef __attribute__((ext_vector_type(4)))  int          v4i;

#define D_MODEL 512
#define JDIM    640
#define VOCAB   1024
#define BB      4
#define TT      256
#define UU      64
#define M_TOTAL 65536

// GEMM tiling: 512 threads = 16 wave32s in a 4x4 wave grid; wave = 32Mx32N.
#define MT     128
#define NT     128
#define KC     64                 // K elements staged per chunk
#define NCHUNK (JDIM / KC)        // 10
#define ROWB   144                // 64 f16 = 128B data + 16B TDM pad (bank-conflict-free, 16B aligned)
#define TILEB  (MT * ROWB)        // 18432 B per staging buffer

// ---------------------------------------------------------------------------
// Small projections: Y[r][j] = dot(X[r,:], W[j,:]) + bias[j]  (f32)
// ---------------------------------------------------------------------------
__global__ void proj_kernel(const float* __restrict__ X,
                            const float* __restrict__ W,
                            const float* __restrict__ bias,
                            float* __restrict__ Y, int R) {
  int idx = blockIdx.x * blockDim.x + threadIdx.x;
  if (idx >= R * JDIM) return;
  int r = idx / JDIM;
  int j = idx - r * JDIM;
  const float4* x = (const float4*)(X + (size_t)r * D_MODEL);
  const float4* w = (const float4*)(W + (size_t)j * D_MODEL);
  float acc = 0.f;
#pragma unroll 4
  for (int k = 0; k < D_MODEL / 4; ++k) {
    float4 a = x[k], b = w[k];
    acc += a.x * b.x + a.y * b.y + a.z * b.z + a.w * b.w;
  }
  Y[idx] = acc + bias[j];
}

__global__ void cvt_kernel(const float* __restrict__ W,
                           _Float16* __restrict__ Wh, int n) {
  int i = blockIdx.x * blockDim.x + threadIdx.x;
  if (i < n) Wh[i] = (_Float16)W[i];
}

// ---------------------------------------------------------------------------
// Z[m][j] = tanh(E[bt][j] + P[b*U+u][j])  computed exactly once (f16, 84MB -> L2)
// ---------------------------------------------------------------------------
__global__ void zgen_kernel(const float* __restrict__ E,
                            const float* __restrict__ P,
                            _Float16* __restrict__ Z) {
  int m  = blockIdx.x;           // 0..65535 flattened (b,t,u)
  int u  = m & (UU - 1);
  int bt = m >> 6;               // b*T + t
  int b  = m >> 14;
  const float* e = E + (size_t)bt * JDIM;
  const float* p = P + (size_t)(b * UU + u) * JDIM;
  _Float16* z = Z + (size_t)m * JDIM;
  for (int j = threadIdx.x; j < JDIM; j += 256)
    z[j] = (_Float16)tanhf(e[j] + p[j]);
}

// ---------------------------------------------------------------------------
// Tile staging: 2D tile (MT rows x KC cols, row stride JDIM) -> padded LDS.
// Preferred path: CDNA5 Tensor Data Mover (tensor_load_to_lds + TENSORcnt).
// ---------------------------------------------------------------------------
#if __has_builtin(__builtin_amdgcn_tensor_load_to_lds)
#define HAS_TDM 1
#else
#define HAS_TDM 0
#endif

#if HAS_TDM
#if defined(__clang_major__) && __clang_major__ >= 23
#define TENSOR_LOAD(g0, g1, g2, g3) \
  __builtin_amdgcn_tensor_load_to_lds(g0, g1, g2, g3, (v8i){0,0,0,0,0,0,0,0}, 0)
#else
#define TENSOR_LOAD(g0, g1, g2, g3) \
  __builtin_amdgcn_tensor_load_to_lds(g0, g1, g2, g3, 0)
#endif

static __device__ inline void tdm_issue(uint32_t lds_addr, const _Float16* gsrc) {
  uint64_t ga = (uint64_t)(uintptr_t)gsrc;
  // ---- D# group 0: count=1 | lds_addr | global_addr[56:0] | type=2 ----
  uint32_t g00 = __builtin_amdgcn_readfirstlane(1u);
  uint32_t g01 = __builtin_amdgcn_readfirstlane(lds_addr);
  uint32_t g02 = __builtin_amdgcn_readfirstlane((uint32_t)ga);
  uint32_t g03 = __builtin_amdgcn_readfirstlane(
      ((uint32_t)(ga >> 32) & 0x01FFFFFFu) | (2u << 30));
  // ---- D# group 1 ----
  // data_size=1 (2B) | pad_enable | pad_interval=4 (32 dwords=128B) | pad_amount=3 (4 dwords=16B)
  uint32_t d0 = (1u << 16) | (1u << 20) | (4u << 22) | (3u << 25);
  uint32_t dim0 = KC, dim1 = MT;                 // tensor dims == tile dims
  uint32_t d1 = (dim0 & 0xFFFFu) << 16;          // tensor_dim0[15:0] at bits 48..63
  uint32_t d2 = (dim0 >> 16) | ((dim1 & 0xFFFFu) << 16);
  uint32_t d3 = (dim1 >> 16) | ((uint32_t)KC << 16);   // tile_dim0 = KC
  uint32_t d4 = (uint32_t)MT;                          // tile_dim1 = MT, tile_dim2 = 0
  uint32_t d5 = (uint32_t)JDIM;                        // tensor_dim0_stride (row stride, elems)
  v4u g0 = { g00, g01, g02, g03 };
  v8i g1 = { (int)d0, (int)d1, (int)d2, (int)d3, (int)d4, (int)d5, 0, 0 };
  v4i gz = { 0, 0, 0, 0 };
  TENSOR_LOAD(g0, g1, gz, gz);
}
#endif

// Stage one MTxKC tile into LDS (padded ROWB layout).
static __device__ inline void stage_tile(char* ldsptr, const _Float16* gsrc, int tid) {
#if HAS_TDM
  if ((tid >> 5) == 0)   // one wave issues the DMA; EXEC is ignored by TDM
    tdm_issue((uint32_t)(uintptr_t)ldsptr, gsrc);
#else
  // Fallback: cooperative 16B copies into identical padded layout.
  for (int c = tid; c < MT * 8; c += 512) {
    int r = c >> 3, k8 = c & 7;
    *(v8h*)(ldsptr + r * ROWB + k8 * 16) =
        *(const v8h*)(gsrc + (size_t)r * JDIM + k8 * 8);
  }
#endif
}

static __device__ inline v16h ldfrag(const char* p) {
  v8h lo = *(const v8h*)(p);
  v8h hi = *(const v8h*)(p + 32);
  v16h f;
#pragma unroll
  for (int i = 0; i < 8; ++i) { f[i] = lo[i]; f[8 + i] = hi[i]; }
  return f;
}

#define WMMA(a, b, c) \
  __builtin_amdgcn_wmma_f32_16x16x32_f16(false, a, false, b, (short)0, c, false, false)

// ---------------------------------------------------------------------------
// Main GEMM: out[m, v] = sum_j Z[m][j] * Wh[v][j] + out_b[v]
// Block: 512 threads, 128Mx128N tile, K chunks of 64, double-buffered TDM staging.
// ---------------------------------------------------------------------------
__global__ __launch_bounds__(512)
void gemm_wmma_kernel(const _Float16* __restrict__ Z,    // (M_TOTAL, J)
                      const _Float16* __restrict__ Wh,   // (V, J)
                      const float* __restrict__ out_b,   // (V,)
                      float* __restrict__ out) {         // (M_TOTAL, V)
  __shared__ __align__(16) char lds[4 * TILEB];  // A0 A1 B0 B1 = 72KB

  const int tid   = threadIdx.x;
  const int mbase = blockIdx.y * MT;
  const int nbase = blockIdx.x * NT;

  const _Float16* Ztile = Z  + (size_t)mbase * JDIM;
  const _Float16* Wtile = Wh + (size_t)nbase * JDIM;

  // Preload chunk 0 into buffers A0/B0
  stage_tile(lds + 0 * TILEB, Ztile, tid);
  stage_tile(lds + 2 * TILEB, Wtile, tid);

  const int wave  = tid >> 5;
  const int lane  = tid & 31;
  const int wm    = wave >> 2;       // 0..3  (M wave group)
  const int wn    = wave & 3;        // 0..3  (N wave group)
  const int row   = lane & 15;       // fragment row / column index
  const int khalf = lane >> 4;       // lanes 16-31 hold the +8/+24 K-halves

  v8f acc00 = {}, acc01 = {}, acc10 = {}, acc11 = {};

  for (int ch = 0; ch < NCHUNK; ++ch) {
    if (ch + 1 < NCHUNK) {           // prefetch next chunk into the other buffer
      const int nb = (ch + 1) & 1;
      stage_tile(lds + nb * TILEB,       Ztile + (size_t)(ch + 1) * KC, tid);
      stage_tile(lds + (2 + nb) * TILEB, Wtile + (size_t)(ch + 1) * KC, tid);
    }
#if HAS_TDM
    if (wave == 0) {                 // wait until only the just-issued 2 remain
      if (ch + 1 < NCHUNK) __builtin_amdgcn_s_wait_tensorcnt(2);
      else                 __builtin_amdgcn_s_wait_tensorcnt(0);
    }
#endif
    __syncthreads();                 // staged data visible to all waves

    const char* Ab = lds + ((ch & 1) ? TILEB : 0);
    const char* Bb = lds + 2 * TILEB + ((ch & 1) ? TILEB : 0);
    const char* Ar = Ab + (wm * 32 + row) * ROWB + khalf * 16;
    const char* Br = Bb + (wn * 32 + row) * ROWB + khalf * 16;

#pragma unroll
    for (int s = 0; s < 2; ++s) {    // two 16x16x32 K-steps per 64-chunk
      const int kb = s * 64;
      v16h a0 = ldfrag(Ar + kb);
      v16h a1 = ldfrag(Ar + 16 * ROWB + kb);
      v16h b0 = ldfrag(Br + kb);
      v16h b1 = ldfrag(Br + 16 * ROWB + kb);
      acc00 = WMMA(a0, b0, acc00);
      acc01 = WMMA(a0, b1, acc01);
      acc10 = WMMA(a1, b0, acc10);
      acc11 = WMMA(a1, b1, acc11);
    }
    __syncthreads();                 // done reading before buffer is re-staged
  }

  // ---- Epilogue: bias + non-temporal store (don't evict Z/W from L2) ----
  const int rbase = mbase + wm * 32 + khalf * 8;
  const int nc0   = nbase + wn * 32 + row;
  const int nc1   = nc0 + 16;
  const float bv0 = out_b[nc0];
  const float bv1 = out_b[nc1];
#pragma unroll
  for (int v = 0; v < 8; ++v) {
    size_t r0 = (size_t)(rbase + v) * VOCAB;
    size_t r1 = (size_t)(rbase + 16 + v) * VOCAB;
    __builtin_nontemporal_store(acc00[v] + bv0, out + r0 + nc0);
    __builtin_nontemporal_store(acc01[v] + bv1, out + r0 + nc1);
    __builtin_nontemporal_store(acc10[v] + bv0, out + r1 + nc0);
    __builtin_nontemporal_store(acc11[v] + bv1, out + r1 + nc1);
  }
}

// ---------------------------------------------------------------------------
extern "C" void kernel_launch(void* const* d_in, const int* in_sizes, int n_in,
                              void* d_out, int out_size, void* d_ws, size_t ws_size,
                              hipStream_t stream) {
  const float* enc    = (const float*)d_in[0];
  const float* pred   = (const float*)d_in[1];
  const float* enc_w  = (const float*)d_in[2];
  const float* enc_b  = (const float*)d_in[3];
  const float* pred_w = (const float*)d_in[4];
  const float* pred_b = (const float*)d_in[5];
  const float* out_w  = (const float*)d_in[6];
  const float* out_b  = (const float*)d_in[7];
  float* out = (float*)d_out;

  char* ws = (char*)d_ws;
  const size_t eBytes = (size_t)BB * TT * JDIM * sizeof(float);     // 2.62 MB
  const size_t pBytes = (size_t)BB * UU * JDIM * sizeof(float);     // 0.66 MB
  const size_t wBytes = (size_t)VOCAB * JDIM * sizeof(_Float16);    // 1.31 MB
  float*    E  = (float*)ws;
  float*    P  = (float*)(ws + eBytes);
  _Float16* Wh = (_Float16*)(ws + eBytes + pBytes);
  _Float16* Zs = (_Float16*)(ws + eBytes + pBytes + wBytes);        // 83.9 MB

  const int nE = BB * TT * JDIM;
  proj_kernel<<<(nE + 255) / 256, 256, 0, stream>>>(enc, enc_w, enc_b, E, BB * TT);
  const int nP = BB * UU * JDIM;
  proj_kernel<<<(nP + 255) / 256, 256, 0, stream>>>(pred, pred_w, pred_b, P, BB * UU);
  const int nW = VOCAB * JDIM;
  cvt_kernel<<<(nW + 255) / 256, 256, 0, stream>>>(out_w, Wh, nW);

  zgen_kernel<<<M_TOTAL, 256, 0, stream>>>(E, P, Zs);

  dim3 grid(VOCAB / NT, M_TOTAL / MT);   // (8, 512)
  gemm_wmma_kernel<<<grid, 512, 0, stream>>>(Zs, Wh, out_b, out);
}
